// SpikingNeuralNetwork_51805895524805
// MI455X (gfx1250) — matrix operand order repaired and checked by reading, not compiled, not run on hardware
//
#include <hip/hip_runtime.h>

typedef __attribute__((ext_vector_type(16))) _Float16 v16h;
typedef __attribute__((ext_vector_type(8)))  _Float16 v8h;
typedef __attribute__((ext_vector_type(8)))  float    v8f;

#define B_ 256
#define T_ 128
#define I_ 512
#define H_ 1024
#define O_ 512
#define ALPHA 0.05f   /* DT/TAU = 1/20 */

// ---- workspace layout (in _Float16 elements) ----
#define WS_WE 0
#define WS_W1 (512*1024)
#define WS_W2 (WS_W1 + 1024*1024)
#define WS_WO (WS_W2 + 1024*1024)

// ---- output layout (in float elements), reference return order ----
static constexpr size_t OUT_LAST = 0;                               // [B,O]
static constexpr size_t OUT_S1   = (size_t)B_*O_;                   // [T,B,H]
static constexpr size_t OUT_S2   = OUT_S1 + (size_t)T_*B_*H_;       // [T,B,H]
static constexpr size_t OUT_SO   = OUT_S2 + (size_t)T_*B_*H_;       // [T,B,O]

// LDS-only workgroup barrier: order LDS ops (dscnt) but let global spike-history
// stores drain in the background (no storecnt stall on the critical chain).
__device__ __forceinline__ void wg_barrier_lds() {
    asm volatile("s_wait_dscnt 0x0\n\t"
                 "s_barrier_signal -1\n\t"
                 "s_barrier_wait -1" ::: "memory");
}

// -------------------------------------------------------------------------
// Weight converter: f32 row-major [K,N] -> f16 WMMA-B-operand swizzled tiles.
// Tile (nt,kt) = 512 halves; lane L holds column n = nt*16 + (L&15),
// half j holds k = kt*32 + 16*(L>>4) + j   (contiguous k order used here, same as the A fetch).
// -------------------------------------------------------------------------
__global__ void snn_convert_w(const float* __restrict__ W, _Float16* __restrict__ dst,
                              int K, int N) {
    const int KT = K >> 5;
    const size_t total = (size_t)K * N;
    for (size_t idx = (size_t)blockIdx.x * blockDim.x + threadIdx.x;
         idx < total; idx += (size_t)gridDim.x * blockDim.x) {
        const size_t tile = idx >> 9;
        const int within = (int)(idx & 511);
        const int L = within >> 4, j = within & 15;
        const int nt = (int)(tile / KT), kt = (int)(tile % KT);
        const int n = nt * 16 + (L & 15);
        const int k = kt * 32 + ((L >> 4) << 4) + j;
        dst[idx] = (_Float16)W[(size_t)k * N + n];
    }
}

// -------------------------------------------------------------------------
// A-operand fetch from LDS spike buffer (16 rows x 1024 halves, row stride 1024).
// A 16x32 k order used here (contiguous, same as B tiles): lane L (m=L&15, g=L>>4):
//   halves 0..7  = K[k0 + 16g .. +7], halves 8..15 = K[k0 + 16g + 8 .. +7]
// -> two 16B ds_load_b128 per tile.
// -------------------------------------------------------------------------
__device__ __forceinline__ v16h snn_load_a(const _Float16* s, int m, int g, int k0) {
    const v8h lo = *(const v8h*)(s + m * 1024 + k0 + 16 * g);
    const v8h hi = *(const v8h*)(s + m * 1024 + k0 + 16 * g + 8);
    return __builtin_shufflevector(lo, hi, 0,1,2,3,4,5,6,7,8,9,10,11,12,13,14,15);
}

// GEMM for one wave: NT consecutive 16-col N-tiles, K = KT*32.
// unroll(1): keeps only a + NT transient B tiles live -> no scratch spills.
template<int NT>
__device__ __forceinline__ void snn_gemm(const _Float16* sIn, int KT,
                                         const _Float16* __restrict__ W,
                                         int wave, int lane, v8f acc[NT]) {
    const int g = lane >> 4, m = lane & 15;
    const int nt0 = wave * NT;
#pragma unroll 1
    for (int kt = 0; kt < KT; ++kt) {
        const v16h a = snn_load_a(sIn, m, g, kt * 32);
        __builtin_prefetch(W + (((size_t)(nt0 * KT + kt + 1)) << 9) + lane * 16, 0, 1);
#pragma unroll
        for (int q = 0; q < NT; ++q) {
            const v16h bm = *(const v16h*)(W + (((size_t)((nt0 + q) * KT + kt)) << 9) + lane * 16);
            acc[q] = __builtin_amdgcn_wmma_f32_16x16x32_f16(
                false, a, false, bm, (short)0, acc[q], false, false);
        }
    }
}

// -------------------------------------------------------------------------
// Persistent SNN kernel: 16 workgroups x 512 threads (16 waves).
// WG owns batch rows [16*blockIdx.x, +16); loops all T timesteps & 4 layers.
// Membrane potentials live in LDS (224KB of the 320KB WGP pool), stored in a
// per-lane-private v8f layout so each tile's v is 2 ds_load_b128/2 ds_store_b128.
// -------------------------------------------------------------------------
__global__ __launch_bounds__(512, 1)
void snn_kernel(const float* __restrict__ x,
                const float* __restrict__ be, const float* __restrict__ b1,
                const float* __restrict__ b2, const float* __restrict__ bo,
                const _Float16* __restrict__ ws,
                float* __restrict__ out) {
    __shared__ _Float16 sA[16 * 1024];                 // 32 KB ping
    __shared__ _Float16 sB[16 * 1024];                 // 32 KB pong
    __shared__ float    vL[16 * 1024 * 2 + 16 * 512];  // 160 KB: v1, v2, vo

    // per-lane-private vector layout: tile (wave*NT+q) -> 32 lanes x v8f
    float* v1L = vL;                    // 64 tiles
    float* v2L = vL + 16 * 1024;        // 64 tiles
    float* voL = vL + 32 * 1024;        // 32 tiles

    const int tid  = threadIdx.x;
    const int lane = tid & 31;
    const int wave = tid >> 5;          // 0..15
    const int g    = lane >> 4;
    const int m    = lane & 15;
    const int b0   = blockIdx.x * 16;

    const _Float16* wWe = ws + WS_WE;
    const _Float16* wW1 = ws + WS_W1;
    const _Float16* wW2 = ws + WS_W2;
    const _Float16* wWo = ws + WS_WO;

    // zero-init membrane potentials (reference starts from v=0)
    for (int i = tid; i < 16 * 1024 * 2 + 16 * 512; i += 512) vL[i] = 0.0f;

    // per-lane biases (n is fixed per lane within a tile)
    float beL[4], b1L[4], b2L[4], boL[2];
#pragma unroll
    for (int q = 0; q < 4; ++q) {
        const int n = wave * 64 + q * 16 + m;
        beL[q] = be[n]; b1L[q] = b1[n]; b2L[q] = b2[n];
    }
#pragma unroll
    for (int q = 0; q < 2; ++q) boL[q] = bo[wave * 32 + q * 16 + m];

    wg_barrier_lds();

#pragma unroll 1
    for (int t = 0; t < T_; ++t) {
        // ---- stage x_t into LDS as f16 (layer-0 A operand) ----
#pragma unroll 1
        for (int i = tid; i < 16 * I_; i += 512) {
            const int row = i >> 9, col = i & 511;
            sA[row * 1024 + col] = (_Float16)x[((size_t)(b0 + row) * T_ + t) * I_ + col];
        }
        wg_barrier_lds();

        // ---- layer 0: h = relu(x@We+be); LIF(v1) -> s1 (to sB + S1) ----
        {
            v8f acc[4] = {};
            snn_gemm<4>(sA, I_ / 32, wWe, wave, lane, acc);
#pragma unroll
            for (int q = 0; q < 4; ++q) {
                const int n = wave * 64 + q * 16 + m;
                float* vp = v1L + (size_t)((wave * 4 + q) * 32 + lane) * 8;
                v8f v = *(v8f*)vp;
#pragma unroll
                for (int r = 0; r < 8; ++r) {
                    const int row = r + 8 * g;
                    const float h = fmaxf(acc[q][r] + beL[q], 0.0f);
                    float vv = v[r] + (h - v[r]) * ALPHA;
                    const bool fire = (vv >= 1.0f);
                    const float sp = fire ? 1.0f : 0.0f;
                    v[r] = fire ? 0.0f : vv;
                    out[OUT_S1 + ((size_t)t * B_ + b0 + row) * H_ + n] = sp;
                    sB[row * 1024 + n] = (_Float16)sp;
                }
                *(v8f*)vp = v;
            }
        }
        wg_barrier_lds();

        // ---- layer 1: s1@W1+b1; LIF(v2) -> s2 (to sA + S2) ----
        {
            v8f acc[4] = {};
            snn_gemm<4>(sB, H_ / 32, wW1, wave, lane, acc);
#pragma unroll
            for (int q = 0; q < 4; ++q) {
                const int n = wave * 64 + q * 16 + m;
                float* vp = v2L + (size_t)((wave * 4 + q) * 32 + lane) * 8;
                v8f v = *(v8f*)vp;
#pragma unroll
                for (int r = 0; r < 8; ++r) {
                    const int row = r + 8 * g;
                    const float h = acc[q][r] + b1L[q];
                    float vv = v[r] + (h - v[r]) * ALPHA;
                    const bool fire = (vv >= 1.0f);
                    const float sp = fire ? 1.0f : 0.0f;
                    v[r] = fire ? 0.0f : vv;
                    out[OUT_S2 + ((size_t)t * B_ + b0 + row) * H_ + n] = sp;
                    sA[row * 1024 + n] = (_Float16)sp;
                }
                *(v8f*)vp = v;
            }
        }
        wg_barrier_lds();

        // ---- layer 2: h2 = s2@W2+b2 (no neuron) -> f16 into sB ----
        {
            v8f acc[4] = {};
            snn_gemm<4>(sA, H_ / 32, wW2, wave, lane, acc);
#pragma unroll
            for (int q = 0; q < 4; ++q) {
                const int n = wave * 64 + q * 16 + m;
#pragma unroll
                for (int r = 0; r < 8; ++r) {
                    const int row = r + 8 * g;
                    sB[row * 1024 + n] = (_Float16)(acc[q][r] + b2L[q]);
                }
            }
        }
        wg_barrier_lds();

        // ---- layer 3: LIF(vo, h2@Wo+bo) -> SO (and last-step output) ----
        {
            v8f acc[2] = {};
            snn_gemm<2>(sB, H_ / 32, wWo, wave, lane, acc);
#pragma unroll
            for (int q = 0; q < 2; ++q) {
                const int n = wave * 32 + q * 16 + m;
                float* vp = voL + (size_t)((wave * 2 + q) * 32 + lane) * 8;
                v8f v = *(v8f*)vp;
#pragma unroll
                for (int r = 0; r < 8; ++r) {
                    const int row = r + 8 * g;
                    const float h = acc[q][r] + boL[q];
                    float vv = v[r] + (h - v[r]) * ALPHA;
                    const bool fire = (vv >= 1.0f);
                    const float sp = fire ? 1.0f : 0.0f;
                    v[r] = fire ? 0.0f : vv;
                    out[OUT_SO + ((size_t)t * B_ + b0 + row) * O_ + n] = sp;
                    if (t == T_ - 1)
                        out[OUT_LAST + (size_t)(b0 + row) * O_ + n] = sp;
                }
                *(v8f*)vp = v;
            }
        }
        wg_barrier_lds();
    }
}

extern "C" void kernel_launch(void* const* d_in, const int* in_sizes, int n_in,
                              void* d_out, int out_size, void* d_ws, size_t ws_size,
                              hipStream_t stream) {
    (void)in_sizes; (void)n_in; (void)out_size; (void)ws_size;
    const float* x  = (const float*)d_in[0];
    const float* We = (const float*)d_in[1];
    const float* be = (const float*)d_in[2];
    const float* W1 = (const float*)d_in[3];
    const float* b1 = (const float*)d_in[4];
    const float* W2 = (const float*)d_in[5];
    const float* b2 = (const float*)d_in[6];
    const float* Wo = (const float*)d_in[7];
    const float* bo = (const float*)d_in[8];
    _Float16* ws = (_Float16*)d_ws;
    float* out = (float*)d_out;

    // pre-swizzle weights into WMMA-B f16 tiles (6 MB, L2-resident afterwards)
    snn_convert_w<<<1024, 256, 0, stream>>>(We, ws + WS_WE, 512, 1024);
    snn_convert_w<<<2048, 256, 0, stream>>>(W1, ws + WS_W1, 1024, 1024);
    snn_convert_w<<<2048, 256, 0, stream>>>(W2, ws + WS_W2, 1024, 1024);
    snn_convert_w<<<1024, 256, 0, stream>>>(Wo, ws + WS_WO, 1024, 512);

    // single persistent kernel: 16 WGs (one per 16-row batch tile), 16 waves each
    snn_kernel<<<16, 512, 0, stream>>>(x, be, b1, b2, bo, ws, out);
}
